// SphericalEmbedding_80264348828009
// MI455X (gfx1250) — compile-verified, matrix operand order checked by reading, not run on hardware
//
#include <hip/hip_runtime.h>
#include <hip/hip_bf16.h>

typedef __attribute__((ext_vector_type(16))) _Float16 v16h;
typedef __attribute__((ext_vector_type(8)))  _Float16 v8h;
typedef __attribute__((ext_vector_type(8)))  float    v8f;

#define KPAD 96     // 81 harmonics padded to 3 chunks of K=32
#define NCH  2048   // output channels
#define NPTS 32768  // 32*1024 points

// ---------------------------------------------------------------------------
// Pack weights f32[81][2048] -> f16 workspace in WMMA B-fragment order.
// Fragment blocks: b = kc*128 + ntile  (kc in [0,3), ntile in [0,128))
// Within block (512 halves): lane L's v16h is contiguous at lane*16.
// Half h of lane L maps to element (k, n):
//   k = 32*kc + (h/8)*16 + (h%8) + 8*(L>=16)      (A/B 16-bit WMMA layout)
//   n = 16*ntile + (L%16)
// k >= 81 is zero padding.
// ---------------------------------------------------------------------------
__global__ __launch_bounds__(256)
void pack_weights(const float* __restrict__ W, _Float16* __restrict__ P) {
    int idx  = blockIdx.x * 256 + threadIdx.x;      // < 96*2048
    int b    = idx >> 9;                            // fragment block (512 halves)
    int r    = idx & 511;
    int lane = r >> 4;
    int h    = r & 15;
    int kc   = b >> 7;
    int nt   = b & 127;
    int k = 32 * kc + ((h >> 3) << 4) + (h & 7) + ((lane >= 16) ? 8 : 0);
    int n = (nt << 4) + (lane & 15);
    float v = (k < 81) ? W[k * NCH + n] : 0.0f;
    P[idx] = (_Float16)v;
}

// ---------------------------------------------------------------------------
// Fused: harmonics (Legendre + Chebyshev, no trig) -> LDS -> WMMA GEMM.
// Block = 256 threads = 8 wave32. Block tile: 32 rows x 256 cols.
// Wave (wm in [0,2), wn in [0,4)) computes 16 rows x 64 cols = 4 tiles,
// K = 96 in 3 chunks of 32 -> 12 v_wmma_f32_16x16x32_f16 per wave.
// ---------------------------------------------------------------------------
__global__ __launch_bounds__(256)
void sph_gemm(const float* __restrict__ pos,
              const _Float16* __restrict__ Wp,
              float* __restrict__ out) {
    __shared__ _Float16 Alds[32 * KPAD];   // 6 KB, rows 16B-aligned (192 B/row)

    const int tid = threadIdx.x;

    // ---- producers: 32 threads compute one point's 81 harmonics each ----
    if (tid < 32) {
        const int gm = blockIdx.y * 32 + tid;
        const float x = pos[gm * 3 + 0];
        const float y = pos[gm * 3 + 1];
        const float z = pos[gm * 3 + 2];
        const float rn  = sqrtf(x * x + y * y + z * z) + 1e-8f;
        float ct = z / rn;
        ct = fminf(1.0f, fmaxf(-1.0f, ct));
        const float st  = sqrtf(fmaxf(0.0f, 1.0f - ct * ct));
        const float rxy = sqrtf(x * x + y * y);
        const float cphi = (rxy > 0.0f) ? (x / rxy) : 1.0f;   // cos(atan2(y,x))

        // Associated Legendre P_l^m(ct), Condon-Shortley phase (scipy).
        float P[9][9];
        #pragma unroll
        for (int m = 0; m <= 8; ++m) {
            float pmm = (m == 0) ? 1.0f
                                 : (-(float)(2 * m - 1) * st * P[m - 1][m - 1]);
            P[m][m] = pmm;
            if (m + 1 <= 8) P[m + 1][m] = (float)(2 * m + 1) * ct * pmm;
            #pragma unroll
            for (int l = m + 2; l <= 8; ++l)
                P[l][m] = ((float)(2 * l - 1) * ct * P[l - 1][m]
                           - (float)(l + m - 1) * P[l - 2][m]) / (float)(l - m);
        }
        // cos(m*phi) Chebyshev recurrence
        float cm[9];
        cm[0] = 1.0f; cm[1] = cphi;
        #pragma unroll
        for (int m = 2; m <= 8; ++m) cm[m] = 2.0f * cphi * cm[m - 1] - cm[m - 2];

        int j = 0;
        #pragma unroll
        for (int l = 0; l <= 8; ++l) {
            #pragma unroll
            for (int m = -l; m <= l; ++m) {
                const int am = (m < 0) ? -m : m;
                float ratio = 1.0f;                     // (l-am)! / (l+am)!
                #pragma unroll
                for (int i = l - am + 1; i <= l + am; ++i) ratio /= (float)i;
                const float norm =
                    sqrtf((float)(2 * l + 1) * 0.07957747154594767f * ratio);
                float Y = norm * P[l][am] * cm[am];
                if (m < 0 && (am & 1)) Y = -Y;
                Alds[tid * KPAD + j] = (_Float16)Y;
                ++j;
            }
        }
        #pragma unroll
        for (int p = 81; p < KPAD; ++p) Alds[tid * KPAD + p] = (_Float16)0.0f;
    }
    __syncthreads();

    // ---- WMMA GEMM ----
    const int lane  = tid & 31;
    const int wave  = tid >> 5;
    const int wm    = wave >> 2;        // 0..1  (row half of block tile)
    const int wn    = wave & 3;         // 0..3  (64-col slab)
    const int rlane = lane & 15;
    const int khalf = (lane >= 16) ? 8 : 0;   // hi lanes carry K+8

    v8f acc0 = {}, acc1 = {}, acc2 = {}, acc3 = {};

    #pragma unroll
    for (int kc = 0; kc < 3; ++kc) {
        // A fragment from LDS: halves 0..7 = K kbase..+7, halves 8..15 = K+16..+23
        const _Float16* ap = &Alds[(wm * 16 + rlane) * KPAD + kc * 32 + khalf];
        v16h a;
        *(v8h*)&a       = *(const v8h*)ap;          // 16B aligned
        *((v8h*)&a + 1) = *(const v8h*)(ap + 16);   // +32 bytes

        const int ntbase = blockIdx.x * 16 + wn * 4;
        const _Float16* bbase = Wp + (size_t)(kc * 128) * 512 + (size_t)lane * 16;

        v16h b0 = *(const v16h*)(bbase + (size_t)(ntbase + 0) * 512);
        v16h b1 = *(const v16h*)(bbase + (size_t)(ntbase + 1) * 512);
        v16h b2 = *(const v16h*)(bbase + (size_t)(ntbase + 2) * 512);
        v16h b3 = *(const v16h*)(bbase + (size_t)(ntbase + 3) * 512);

        acc0 = __builtin_amdgcn_wmma_f32_16x16x32_f16(false, a, false, b0,
                                                      (short)0, acc0, false, false);
        acc1 = __builtin_amdgcn_wmma_f32_16x16x32_f16(false, a, false, b1,
                                                      (short)0, acc1, false, false);
        acc2 = __builtin_amdgcn_wmma_f32_16x16x32_f16(false, a, false, b2,
                                                      (short)0, acc2, false, false);
        acc3 = __builtin_amdgcn_wmma_f32_16x16x32_f16(false, a, false, b3,
                                                      (short)0, acc3, false, false);
    }

    // D layout: VGPR v -> row v + 8*(lane>=16), col = lane%16
    const size_t m0 = (size_t)blockIdx.y * 32 + wm * 16 + ((lane >= 16) ? 8 : 0);
    const size_t n0 = (size_t)blockIdx.x * 256 + wn * 64 + rlane;
    float* obase = out + m0 * NCH + n0;
    #pragma unroll
    for (int v = 0; v < 8; ++v) {
        obase[(size_t)v * NCH +  0] = acc0[v];
        obase[(size_t)v * NCH + 16] = acc1[v];
        obase[(size_t)v * NCH + 32] = acc2[v];
        obase[(size_t)v * NCH + 48] = acc3[v];
    }
}

extern "C" void kernel_launch(void* const* d_in, const int* in_sizes, int n_in,
                              void* d_out, int out_size, void* d_ws, size_t ws_size,
                              hipStream_t stream) {
    const float* positions = (const float*)d_in[0];   // [32,1024,3] f32
    const float* weights   = (const float*)d_in[1];   // [81,2048]   f32
    float* out = (float*)d_out;                       // [32,1024,2048] f32
    _Float16* Wp = (_Float16*)d_ws;                   // 96*2048 halves = 384 KB

    pack_weights<<<(KPAD * NCH) / 256, 256, 0, stream>>>(weights, Wp);

    dim3 grid(NCH / 256, NPTS / 32);                  // (8, 1024)
    sph_gemm<<<grid, 256, 0, stream>>>(positions, Wp, out);
}
